// ParallelLSTMs_72069551227363
// MI455X (gfx1250) — matrix operand order repaired and checked by reading, not compiled, not run on hardware
//
#include <hip/hip_runtime.h>
#include <hip/hip_bf16.h>
#include <math.h>

#define BB   256     // batch
#define TT   256     // timesteps
#define II   128     // input dim
#define HH   512     // hidden dim
#define OO   6       // output dim per branch
#define NBR  5       // branches
#define G4H  2048    // 4*H gate width
#define KTOT 640     // I + H combined K
#define NKT  20      // KTOT / 32
#define LDSK 648     // padded K pitch (bank-conflict avoidance)
#define BT   16      // batch tile per workgroup pair
#define NSPL 2       // N-split: WGs cooperating on one batch tile
#define HHS  256     // hidden units owned per WG (HH / NSPL)
#define NBT  16      // batch tiles (BB / BT)

// workspace layout (bytes)
#define WC_BYTES   ((size_t)NBR * G4H * KTOT * 2)                // 13,107,200
#define HX_OFF     WC_BYTES
#define HX_BYTES   ((size_t)NBR * NBT * NSPL * 2 * BT * HHS * 2) // parity-dbl
#define FLAG_OFF   (HX_OFF + HX_BYTES)
#define NFLAGS     (NBR * NBT * NSPL)

#define AS3 __attribute__((address_space(3)))

#if __has_builtin(__builtin_amdgcn_global_load_async_to_lds_b128) && \
    __has_builtin(__builtin_amdgcn_global_store_async_from_lds_b128)
#define USE_ASYNC_XCH 1
#else
#define USE_ASYNC_XCH 0
#endif

typedef __attribute__((ext_vector_type(16))) __bf16       v16bf;
typedef __attribute__((ext_vector_type(2)))  __bf16       v2bf;
typedef __attribute__((ext_vector_type(8)))  float        v8f;
typedef __attribute__((ext_vector_type(4)))  unsigned int v4u;
typedef int v4i_ __attribute__((vector_size(16)));  // async builtin arg type

union AB32 { v4u u[2]; v16bf v; };

__device__ __forceinline__ unsigned short f2bf(float f) {
  unsigned int u = __float_as_uint(f);
  u += 0x7FFFu + ((u >> 16) & 1u);           // round-to-nearest-even
  return (unsigned short)(u >> 16);
}
__device__ __forceinline__ float bf2f(unsigned short s) {
  return __uint_as_float(((unsigned int)s) << 16);
}
__device__ __forceinline__ unsigned int pack_bf16x2(float lo, float hi) {
#if __has_builtin(__builtin_amdgcn_cvt_pk_bf16_f32)
  union { v2bf v; unsigned int u; } cv;
  cv.v = __builtin_amdgcn_cvt_pk_bf16_f32(lo, hi);
  return cv.u;
#else
  return (unsigned int)f2bf(lo) | ((unsigned int)f2bf(hi) << 16);
#endif
}
// gfx1250 hardware v_tanh_f32; fallback avoids the IEEE-divide expansion.
__device__ __forceinline__ float ftanh(float x) {
#if __has_builtin(__builtin_amdgcn_tanhf)
  return __builtin_amdgcn_tanhf(x);
#elif __has_builtin(__builtin_amdgcn_tanh_f32)
  return __builtin_amdgcn_tanh_f32(x);
#else
  x = fminf(15.0f, fmaxf(-15.0f, x));
  float e = __expf(2.0f * x);
  return (e - 1.0f) * __builtin_amdgcn_rcpf(e + 1.0f);
#endif
}
__device__ __forceinline__ float fsigmoid(float x) {
  return fmaf(0.5f, ftanh(0.5f * x), 0.5f);   // 0.5*tanh(x/2)+0.5
}
__device__ __forceinline__ void wait_async0() {
#if __has_builtin(__builtin_amdgcn_s_wait_asynccnt)
  __builtin_amdgcn_s_wait_asynccnt(0);
#else
  asm volatile("s_wait_asynccnt 0" ::: "memory");
#endif
}

// fp32 W_ih/W_hh -> combined bf16 Wc[n][N=2048][K=640] (K contiguous),
// plus zero the per-pair handshake flags for this launch.
__global__ void prep_weights_kernel(const float* __restrict__ W_ih,
                                    const float* __restrict__ W_hh,
                                    unsigned short* __restrict__ Wc,
                                    int* __restrict__ flags) {
  size_t idx = (size_t)blockIdx.x * 256 + threadIdx.x;
  if (idx < NFLAGS) flags[idx] = 0;
  if (idx >= (size_t)NBR * G4H * KTOT) return;
  int    k  = (int)(idx % KTOT);
  size_t gn = idx / KTOT;
  int    g  = (int)(gn % G4H);
  int    n  = (int)(gn / G4H);
  float v = (k < II) ? W_ih[((size_t)n * G4H + g) * II + k]
                     : W_hh[((size_t)n * G4H + g) * HH + (k - II)];
  Wc[idx] = f2bf(v);
}

__global__ __launch_bounds__(256) void lstm_branch_kernel(
    const float* __restrict__ x,
    const unsigned short* __restrict__ Wc,
    unsigned short* __restrict__ hxch,  // [fidx][parity][BT][HHS] bf16
    int* __restrict__ flags,            // [fidx] monotonic step counters
    const float* __restrict__ b_ih, const float* __restrict__ b_hh,
    const float* __restrict__ W_mu, const float* __restrict__ b_mu,
    const float* __restrict__ W_sig, const float* __restrict__ b_sig,
    float* __restrict__ out)
{
  __shared__ unsigned short smem[2][BT][LDSK];  // ping-pong [x | h] bf16

  const int tid  = threadIdx.x;
  const int lane = tid & 31;
  const int wv   = tid >> 5;      // wave 0..7
  const int lh   = lane >> 4;     // lane half (WMMA layout)
  const int l16  = lane & 15;

  const int bt = blockIdx.x >> 1; // batch tile
  const int hs = blockIdx.x & 1;  // N-split half (owns hidden units hs*256..)
  const int n  = blockIdx.y;      // branch
  const int b0 = bt * BT;

  const int fidx  = ((n * NBT + bt) * NSPL) + hs;
  const int pfidx = ((n * NBT + bt) * NSPL) + (hs ^ 1);
  unsigned short* hx_own  = hxch + (size_t)fidx  * 2 * BT * HHS;
  unsigned short* hx_part = hxch + (size_t)pfidx * 2 * BT * HHS;
  int* flag_own  = flags + fidx;
  int* flag_part = flags + pfidx;

  const unsigned short* wb = Wc + (size_t)n * G4H * KTOT;

  // ---- per-lane gate biases for the 2 hidden tiles this wave owns ----
  float bias[2][4];
#pragma unroll
  for (int jl = 0; jl < 2; ++jl) {
    const int jt = wv * 2 + jl;                 // 0..15 within this WG's half
#pragma unroll
    for (int g = 0; g < 4; ++g) {
      const int col = g * HH + hs * HHS + jt * 16 + l16;
      bias[jl][g] = b_ih[(size_t)n * G4H + col] + b_hh[(size_t)n * G4H + col];
    }
  }

  // ---- zero full h-region of buffer 0 (h_{-1} = 0, both halves) ----
  {
    const v4u z = {0u, 0u, 0u, 0u};
#pragma unroll
    for (int i = 0; i < 4; ++i) {
      const int idx = tid + i * 256;   // 1024 stores of 8 halfs = 16 x 512
      const int row = idx >> 6;
      const int c   = idx & 63;
      *(v4u*)&smem[0][row][II + c * 8] = z;
    }
  }
  // ---- stage x_0 into buffer 0 ----
  {
    const int row = tid >> 4;
    const int k   = (tid & 15) * 8;
    const float* xp = x + ((size_t)(b0 + row) * TT + 0) * II + k;
    unsigned int pk[4];
#pragma unroll
    for (int q = 0; q < 4; ++q) pk[q] = pack_bf16x2(xp[2 * q], xp[2 * q + 1]);
    *(v4u*)&smem[0][row][k] = *(v4u*)pk;
  }

  v8f cst[2];                       // register-resident cell state
#pragma unroll
  for (int jl = 0; jl < 2; ++jl)
#pragma unroll
    for (int r = 0; r < 8; ++r) cst[jl][r] = 0.0f;

  __syncthreads();

  for (int t = 0; t < TT; ++t) {
    const int p  = t & 1;
    const int np = 1 - p;

    // stage x_{t+1} into the other buffer
    if (t + 1 < TT) {
      const int row = tid >> 4;
      const int k   = (tid & 15) * 8;
      const float* xp = x + ((size_t)(b0 + row) * TT + (t + 1)) * II + k;
      unsigned int pk[4];
#pragma unroll
      for (int q = 0; q < 4; ++q) pk[q] = pack_bf16x2(xp[2 * q], xp[2 * q + 1]);
      *(v4u*)&smem[np][row][k] = *(v4u*)pk;
    }

#pragma unroll
    for (int jl = 0; jl < 2; ++jl) {
      const int jt = wv * 2 + jl;

      v8f acc[4];
#pragma unroll
      for (int g = 0; g < 4; ++g)
#pragma unroll
        for (int r = 0; r < 8; ++r) acc[g][r] = bias[jl][g];

#pragma unroll 4
      for (int kt = 0; kt < NKT; ++kt) {
        AB32 af;  // A: row m = l16, two contiguous 16B chunks (WMMA layout)
        const unsigned short* arow = &smem[p][l16][kt * 32 + lh * 8];
        af.u[0] = *(const v4u*)(arow);
        af.u[1] = *(const v4u*)(arow + 16);
#pragma unroll
        for (int g = 0; g < 4; ++g) {
          const int nrow = g * HH + hs * HHS + jt * 16 + l16;
          const unsigned short* brow = wb + (size_t)nrow * KTOT + kt * 32 + lh * 8;
          AB32 bf;
          bf.u[0] = *(const v4u*)(brow);
          bf.u[1] = *(const v4u*)(brow + 16);
          acc[g] = __builtin_amdgcn_wmma_f32_16x16x32_bf16(
              false, af.v, false, bf.v, (short)0, acc[g], false, false);
        }
      }

      // fused gate math (i,f,g,o for same hidden units are in-lane)
#pragma unroll
      for (int r = 0; r < 8; ++r) {
        const float gi = fsigmoid(acc[0][r]);
        const float gf = fsigmoid(acc[1][r]);
        const float gg = ftanh(acc[2][r]);
        const float go = fsigmoid(acc[3][r]);
        const float c  = gf * cst[jl][r] + gi * gg;
        cst[jl][r] = c;
        const float h = go * ftanh(c);
        smem[np][lh * 8 + r][II + hs * HHS + jt * 16 + l16] = f2bf(h);
      }
    }
    __syncthreads();   // own h half complete in LDS

    // ---- export own h half (16 x 256 bf16, coalesced) to L2 ----
    {
      unsigned short* dst = hx_own + (size_t)(t & 1) * BT * HHS;
#pragma unroll
      for (int i = 0; i < 2; ++i) {
        const int idx = tid + i * 256;          // 512 x 16B
        const int row = idx >> 5;
        const int c   = idx & 31;
#if USE_ASYNC_XCH
        __builtin_amdgcn_global_store_async_from_lds_b128(
            (v4i_*)(dst + row * HHS + c * 8),
            (AS3 v4i_*)&smem[np][row][II + hs * HHS + c * 8], 0, 0);
#else
        *(v4u*)(dst + row * HHS + c * 8) =
            *(const v4u*)&smem[np][row][II + hs * HHS + c * 8];
#endif
      }
    }
#if USE_ASYNC_XCH
    wait_async0();      // drain ASYNCcnt (not covered by the fence's counters)
#endif
    __threadfence();
    __syncthreads();
    if (tid == 0)
      __hip_atomic_store(flag_own, t + 1, __ATOMIC_RELEASE,
                         __HIP_MEMORY_SCOPE_AGENT);

    // ---- import partner h half for this step ----
    while (__hip_atomic_load(flag_part, __ATOMIC_ACQUIRE,
                             __HIP_MEMORY_SCOPE_AGENT) <= t) {
      __builtin_amdgcn_s_sleep(1);
    }
    {
      const unsigned short* src = hx_part + (size_t)(t & 1) * BT * HHS;
#pragma unroll
      for (int i = 0; i < 2; ++i) {
        const int idx = tid + i * 256;
        const int row = idx >> 5;
        const int c   = idx & 31;
#if USE_ASYNC_XCH
        __builtin_amdgcn_global_load_async_to_lds_b128(
            (v4i_*)(src + row * HHS + c * 8),
            (AS3 v4i_*)&smem[np][row][II + (hs ^ 1) * HHS + c * 8], 0, 0);
#else
        *(v4u*)&smem[np][row][II + (hs ^ 1) * HHS + c * 8] =
            *(const v4u*)(src + row * HHS + c * 8);
#endif
      }
    }
#if USE_ASYNC_XCH
    wait_async0();
#endif
    __syncthreads();   // buf[np] fully ready (x_{t+1} | h_t both halves)
  }

  // ---- output heads on final h (full 512 cols in smem[TT&1]) ----
  // hs==0 WG writes mu, hs==1 WG writes sigma (both hold full h).
  const int hb = TT & 1;
  if (tid < 96) {
    const int which = hs;
    const int rr = tid / OO;
    const int oo = tid % OO;
    const float* W    = which ? W_sig : W_mu;
    const float* bvec = which ? b_sig : b_mu;
    float s = bvec[(size_t)n * OO + oo];
    const float* wrow = W + ((size_t)n * OO + oo) * HH;
    const unsigned short* hrow = &smem[hb][rr][II];
    for (int k = 0; k < HH; ++k) s += bf2f(hrow[k]) * wrow[k];
    if (which) s = __expf(s);
    out[(size_t)which * BB * NBR * OO + (size_t)(b0 + rr) * NBR * OO
        + (size_t)n * OO + oo] = s;
  }
}

extern "C" void kernel_launch(void* const* d_in, const int* in_sizes, int n_in,
                              void* d_out, int out_size, void* d_ws, size_t ws_size,
                              hipStream_t stream) {
  const float* x     = (const float*)d_in[0];
  const float* W_ih  = (const float*)d_in[1];
  const float* W_hh  = (const float*)d_in[2];
  const float* b_ih  = (const float*)d_in[3];
  const float* b_hh  = (const float*)d_in[4];
  const float* W_mu  = (const float*)d_in[5];
  const float* b_mu  = (const float*)d_in[6];
  const float* W_sig = (const float*)d_in[7];
  const float* b_sig = (const float*)d_in[8];
  float* out = (float*)d_out;

  // workspace: bf16 weights (~12.5MB, L2-resident) + h-exchange + flags
  unsigned short* Wc    = (unsigned short*)d_ws;
  unsigned short* hxch  = (unsigned short*)((char*)d_ws + HX_OFF);
  int*            flags = (int*)((char*)d_ws + FLAG_OFF);

  const size_t total = (size_t)NBR * G4H * KTOT;
  prep_weights_kernel<<<(unsigned)((total + 255) / 256), 256, 0, stream>>>(
      W_ih, W_hh, Wc, flags);

  dim3 grid(NSPL * NBT, NBR);   // 160 workgroups
  lstm_branch_kernel<<<grid, 256, 0, stream>>>(
      x, Wc, hxch, flags, b_ih, b_hh, W_mu, b_mu, W_sig, b_sig, out);
}
